// MultiHeadAttention_88124138979383
// MI455X (gfx1250) — matrix operand order repaired and checked by reading, not compiled, not run on hardware
//
#include <hip/hip_runtime.h>

#define DM     1024
#define NHEADS 16
#define DH     64
#define LSEQ   2048
#define NB     2

typedef __attribute__((ext_vector_type(16))) __bf16 v16bf;
typedef __attribute__((ext_vector_type(8)))  float  v8f;
typedef __attribute__((ext_vector_type(8)))  int    v8i;
typedef __attribute__((ext_vector_type(4)))  int    v4i;

// ---------- helpers ----------

__device__ __forceinline__ unsigned short f2bf(float f) {
  unsigned int u = __builtin_bit_cast(unsigned int, f);
  u += 0x7fffu + ((u >> 16) & 1u);       // round-to-nearest-even
  return (unsigned short)(u >> 16);
}

__device__ __forceinline__ v8f wmma_bf16(v8i a, v8i b, v8f c) {
  return __builtin_amdgcn_wmma_f32_16x16x32_bf16(
      false, __builtin_bit_cast(v16bf, a),
      false, __builtin_bit_cast(v16bf, b),
      (short)0, c, false, false);
}

// A-operand fragment (16x32 bf16): lane%16 = M row; VGPR v<4: K=8h+2v, v>=4: K=16+8h+2(v-4).
__device__ __forceinline__ v8i load_fragA(const unsigned short* rowPtr) {
  const v4i* p = (const v4i*)rowPtr;
  const int h = (threadIdx.x >> 4) & 1;
  v4i lo = p[h];       // K pairs 4h..4h+3
  v4i hi = p[2 + h];   // K pairs 8+4h..
  v8i f;
  f[0] = lo[0]; f[1] = lo[1]; f[2] = lo[2]; f[3] = lo[3];
  f[4] = hi[0]; f[5] = hi[1]; f[6] = hi[2]; f[7] = hi[3];
  return f;
}

// B-operand fragment (32x16 bf16): lane%16 = N col; VGPR v<4: K=8h+2v, v>=4: K=16+8h+2(v-4).
__device__ __forceinline__ v8i load_fragB(const unsigned short* rowPtr) {
  const v4i* p = (const v4i*)rowPtr;
  const int h = (threadIdx.x >> 4) & 1;
  v4i lo = p[h];       // K pairs 4h..4h+3
  v4i hi = p[2 + h];   // K pairs 8+4h..
  v8i f;
  f[0] = lo[0]; f[1] = lo[1]; f[2] = lo[2]; f[3] = lo[3];
  f[4] = hi[0]; f[5] = hi[1]; f[6] = hi[2]; f[7] = hi[3];
  return f;
}

// ---------- conversion / transpose kernels ----------

__global__ void __launch_bounds__(256) cvt_k(const float* __restrict__ in,
                                             unsigned short* __restrict__ out, int n) {
  int i = (blockIdx.x * 256 + threadIdx.x) * 4;
  if (i + 3 < n) {
    float4 v = *(const float4*)(in + i);
    unsigned int lo = (unsigned int)f2bf(v.x) | ((unsigned int)f2bf(v.y) << 16);
    unsigned int hi = (unsigned int)f2bf(v.z) | ((unsigned int)f2bf(v.w) << 16);
    ((unsigned int*)out)[i / 2]     = lo;
    ((unsigned int*)out)[i / 2 + 1] = hi;
  }
}

// W[k][n] fp32 -> Wt[n][k] bf16, with optional constant pre-scale (folds softmax scale into Wq)
__global__ void __launch_bounds__(256) wtcvt_k(const float* __restrict__ W,
                                               unsigned short* __restrict__ Wt,
                                               int Kd, int Nd, float mul) {
  int idx = blockIdx.x * 256 + threadIdx.x;
  if (idx < Kd * Nd) {
    int k = idx / Nd, n = idx - k * Nd;
    Wt[(size_t)n * Kd + k] = f2bf(W[idx] * mul);
  }
}

// V [B*L, H*DH] bf16 -> Vt [B, H, DH, L] bf16
__global__ void __launch_bounds__(256) vtrans_k(const unsigned short* __restrict__ V,
                                                unsigned short* __restrict__ Vt) {
  int idx = blockIdx.x * 256 + threadIdx.x;           // over NB*LSEQ*DM
  int tok = idx >> 10;
  int c   = idx & 1023;
  int h = c >> 6, d = c & 63;
  int b = tok >> 11, l = tok & 2047;
  Vt[(((size_t)(b * NHEADS + h)) * DH + d) * LSEQ + l] = V[idx];
}

// ---------- WMMA GEMM: C[M,N] = A[M,K](bf16) * Bt[N,K](bf16)^T + bias*biasMul ----------
// MODE 0: bf16 out.  MODE 1: fp32 out with residual add (for final projection).
// Double-buffered K-slices so next loads are in flight while current WMMAs issue.
template <int MODE>
__global__ void __launch_bounds__(256) gemm_bf16_k(
    const unsigned short* __restrict__ A, const unsigned short* __restrict__ Bt,
    const float* __restrict__ bias, float biasMul, const float* __restrict__ resid,
    unsigned short* __restrict__ outB, float* __restrict__ outF,
    int M, int N, int K) {
  const int wave = threadIdx.x >> 5;
  const int lane = threadIdx.x & 31;
  const int half = lane >> 4;
  const int mW = blockIdx.y * 128 + (wave & 3) * 32;   // 2 M-tiles per wave
  const int nW = blockIdx.x * 128 + (wave >> 2) * 64;  // 4 N-tiles per wave

  const unsigned short* aPtr[2];
  const unsigned short* bPtr[4];
#pragma unroll
  for (int i = 0; i < 2; ++i) aPtr[i] = A + (size_t)(mW + i * 16 + (lane & 15)) * K;
#pragma unroll
  for (int j = 0; j < 4; ++j) bPtr[j] = Bt + (size_t)(nW + j * 16 + (lane & 15)) * K;

  v8f acc[2][4];
  const v8f vzero = {0.f, 0.f, 0.f, 0.f, 0.f, 0.f, 0.f, 0.f};
#pragma unroll
  for (int i = 0; i < 2; ++i)
#pragma unroll
    for (int j = 0; j < 4; ++j) acc[i][j] = vzero;

  // prologue loads
  v8i fa[2], fb[4];
#pragma unroll
  for (int i = 0; i < 2; ++i) fa[i] = load_fragA(aPtr[i]);
#pragma unroll
  for (int j = 0; j < 4; ++j) fb[j] = load_fragB(bPtr[j]);

  for (int kb = 0; kb < K; kb += 32) {
    const int kn = (kb + 32 < K) ? kb + 32 : 0;   // clamp keeps loads in-bounds
    v8i na[2], nb[4];
#pragma unroll
    for (int i = 0; i < 2; ++i) na[i] = load_fragA(aPtr[i] + kn);
#pragma unroll
    for (int j = 0; j < 4; ++j) nb[j] = load_fragB(bPtr[j] + kn);
    if (kb + 64 < K) {                             // global_prefetch_b8 two slices ahead
      __builtin_prefetch(aPtr[0] + kb + 64, 0, 1);
      __builtin_prefetch(bPtr[0] + kb + 64, 0, 1);
    }
#pragma unroll
    for (int i = 0; i < 2; ++i)
#pragma unroll
      for (int j = 0; j < 4; ++j) acc[i][j] = wmma_bf16(fa[i], fb[j], acc[i][j]);
#pragma unroll
    for (int i = 0; i < 2; ++i) fa[i] = na[i];
#pragma unroll
    for (int j = 0; j < 4; ++j) fb[j] = nb[j];
  }

#pragma unroll
  for (int i = 0; i < 2; ++i) {
#pragma unroll
    for (int j = 0; j < 4; ++j) {
      const int n = nW + j * 16 + (lane & 15);
      const float bv = bias[n] * biasMul;
#pragma unroll
      for (int r = 0; r < 8; ++r) {
        const int m = mW + i * 16 + 8 * half + r;     // C-layout row
        const size_t idx = (size_t)m * N + n;
        const float v = acc[i][j][r] + bv;
        if (MODE == 0) outB[idx] = f2bf(v);
        else           outF[idx] = v + resid[idx];
      }
    }
  }
}

// ---------- causal flash attention (per wave: 16 query rows) ----------
// S^T = K*Q^T so softmax stats are per-lane and P packs in-lane into A-fragments.
// Q was pre-scaled by 0.125*log2(e) -> online softmax runs in the base-2 domain (v_exp_f32).
__global__ void __launch_bounds__(256) flash_attn_k(
    const unsigned short* __restrict__ Q,   // [B*L, DM] bf16 (pre-scaled)
    const unsigned short* __restrict__ Kp,  // [B*L, DM] bf16
    const unsigned short* __restrict__ Vt,  // [B, H, DH, L] bf16
    unsigned short* __restrict__ O) {       // [B*L, DM] bf16
  const int b = blockIdx.z, h = blockIdx.y;
  const int wave  = threadIdx.x >> 5;
  const int lane  = threadIdx.x & 31;
  const int hhalf = lane >> 4;
  const int qBase = blockIdx.x * 128 + wave * 16;

  const unsigned short* Qh = Q  + (size_t)b * LSEQ * DM + h * DH;
  const unsigned short* Kh = Kp + (size_t)b * LSEQ * DM + h * DH;
  const unsigned short* Vh = Vt + ((size_t)(b * NHEADS + h)) * DH * LSEQ;

  // Q as B-operand: lane = query column of Q^T, contiguous d.
  const unsigned short* qRow = Qh + (size_t)(qBase + (lane & 15)) * DM;
  v8i fq0 = load_fragB(qRow);
  v8i fq1 = load_fragB(qRow + 32);

  const v8f vzero = {0.f, 0.f, 0.f, 0.f, 0.f, 0.f, 0.f, 0.f};
  v8f o[4] = {vzero, vzero, vzero, vzero};
  float mrow = -1e30f, lrow = 0.f;
  const int qi = qBase + (lane & 15);
  const int kEnd = qBase + 16;                // causal: keys < kEnd (rounded up per block)

  auto loadK = [&](int kb, v8i* out) {
    int kr0 = kb + (lane & 15);      if (kr0 > LSEQ - 1) kr0 = LSEQ - 1;
    int kr1 = kb + 16 + (lane & 15); if (kr1 > LSEQ - 1) kr1 = LSEQ - 1;
    const unsigned short* k0p = Kh + (size_t)kr0 * DM;
    const unsigned short* k1p = Kh + (size_t)kr1 * DM;
    out[0] = load_fragA(k0p); out[1] = load_fragA(k0p + 32);
    out[2] = load_fragA(k1p); out[3] = load_fragA(k1p + 32);
  };

  v8i fk[4];
  loadK(0, fk);                                // prologue

  for (int kb = 0; kb < kEnd; kb += 32) {
    // issue next K-block loads + this block's V loads up front (in flight during VALU work)
    const int kn = (kb + 32 < kEnd) ? kb + 32 : 0;
    v8i nk[4];
    loadK(kn, nk);
    v8i fv[4];
#pragma unroll
    for (int t = 0; t < 4; ++t)
      fv[t] = load_fragB(Vh + (size_t)(t * 16 + (lane & 15)) * LSEQ + kb);
    __builtin_prefetch(Kh + (size_t)(kb + 64) * DM, 0, 1);

    v8f st0 = vzero, st1 = vzero;               // S^T tiles: m=key, n=q
    st0 = wmma_bf16(fk[0], fq0, st0);
    st0 = wmma_bf16(fk[1], fq1, st0);
    st1 = wmma_bf16(fk[2], fq0, st1);
    st1 = wmma_bf16(fk[3], fq1, st1);
#pragma unroll
    for (int t = 0; t < 4; ++t) fk[t] = nk[t];

    // causal mask only on diagonal blocks (wave-uniform branch)
    if (kb + 31 > qBase) {
      const int keyB = kb + 8 * hhalf;          // vgpr r holds key = keyB + r (tile0), +16 (tile1)
#pragma unroll
      for (int r = 0; r < 8; ++r) {
        st0[r] = (keyB + r      > qi) ? -1e30f : st0[r];
        st1[r] = (keyB + 16 + r > qi) ? -1e30f : st1[r];
      }
    }

    float tmax = -1e30f;
#pragma unroll
    for (int r = 0; r < 8; ++r) tmax = fmaxf(tmax, fmaxf(st0[r], st1[r]));
    tmax = fmaxf(tmax, __shfl_xor(tmax, 16, 32));   // merge the two key-halves
    const float mnew = fmaxf(mrow, tmax);
    const float corr = exp2f(mrow - mnew);
    mrow = mnew;

    float rsum = 0.f;
    unsigned int pk0[4], pk1[4];
#pragma unroll
    for (int v = 0; v < 4; ++v) {
      float e0a = exp2f(st0[2 * v]     - mnew);
      float e0b = exp2f(st0[2 * v + 1] - mnew);
      float e1a = exp2f(st1[2 * v]     - mnew);
      float e1b = exp2f(st1[2 * v + 1] - mnew);
      rsum += e0a + e0b + e1a + e1b;
      pk0[v] = (unsigned int)f2bf(e0a) | ((unsigned int)f2bf(e0b) << 16);
      pk1[v] = (unsigned int)f2bf(e1a) | ((unsigned int)f2bf(e1b) << 16);
    }
    lrow = lrow * corr + (rsum + __shfl_xor(rsum, 16, 32));

    // P packs in-lane into the A-fragment layout (keys = contraction dim)
    v8i fp;
    fp[0] = (int)pk0[0]; fp[1] = (int)pk0[1]; fp[2] = (int)pk0[2]; fp[3] = (int)pk0[3];
    fp[4] = (int)pk1[0]; fp[5] = (int)pk1[1]; fp[6] = (int)pk1[2]; fp[7] = (int)pk1[3];

    // O accumulators are in C-layout (row = 8*hhalf + r): fetch per-row rescale factors
    float corrR[8];
#pragma unroll
    for (int r = 0; r < 8; ++r) corrR[r] = __shfl(corr, 8 * hhalf + r, 32);

#pragma unroll
    for (int t = 0; t < 4; ++t) {               // 4 d-tiles of 16
#pragma unroll
      for (int r = 0; r < 8; ++r) o[t][r] *= corrR[r];
      o[t] = wmma_bf16(fp, fv[t], o[t]);
    }
  }

  const float linv = 1.0f / lrow;
  float lR[8];
#pragma unroll
  for (int r = 0; r < 8; ++r) lR[r] = __shfl(linv, 8 * hhalf + r, 32);

  unsigned short* Ob = O + (size_t)b * LSEQ * DM + h * DH;
#pragma unroll
  for (int t = 0; t < 4; ++t) {
    const int d = t * 16 + (lane & 15);
#pragma unroll
    for (int r = 0; r < 8; ++r) {
      const int tok = qBase + 8 * hhalf + r;
      Ob[(size_t)tok * DM + d] = f2bf(o[t][r] * lR[r]);
    }
  }
}

// ---------- rowwise LayerNorm ----------
__global__ void __launch_bounds__(256) ln_k(const float* __restrict__ Y,
                                            const float* __restrict__ gamma,
                                            const float* __restrict__ beta,
                                            float* __restrict__ out) {
  __shared__ float sSum[256], sSq[256];
  const int row = blockIdx.x;
  const float* y = Y + (size_t)row * DM;
  float s = 0.f, sq = 0.f;
  for (int i = threadIdx.x; i < DM; i += 256) { float v = y[i]; s += v; sq += v * v; }
  sSum[threadIdx.x] = s; sSq[threadIdx.x] = sq;
  __syncthreads();
  for (int st = 128; st > 0; st >>= 1) {
    if (threadIdx.x < st) {
      sSum[threadIdx.x] += sSum[threadIdx.x + st];
      sSq[threadIdx.x]  += sSq[threadIdx.x + st];
    }
    __syncthreads();
  }
  const float mu  = sSum[0] * (1.0f / DM);
  const float var = sSq[0] * (1.0f / DM) - mu * mu;
  const float rstd = rsqrtf(var + 1e-5f);
  for (int i = threadIdx.x; i < DM; i += 256)
    out[(size_t)row * DM + i] = gamma[i] * (y[i] - mu) * rstd + beta[i];
}

// ---------- launch ----------
extern "C" void kernel_launch(void* const* d_in, const int* in_sizes, int n_in,
                              void* d_out, int out_size, void* d_ws, size_t ws_size,
                              hipStream_t stream) {
  const float* x_q  = (const float*)d_in[0];
  const float* x_k  = (const float*)d_in[1];
  const float* x_v  = (const float*)d_in[2];
  // d_in[3] = causal mask (bool) — computed analytically in-kernel
  const float* Wq = (const float*)d_in[4];  const float* bq = (const float*)d_in[5];
  const float* Wk = (const float*)d_in[6];  const float* bk = (const float*)d_in[7];
  const float* Wv = (const float*)d_in[8];  const float* bv = (const float*)d_in[9];
  const float* Wo = (const float*)d_in[10]; const float* bo = (const float*)d_in[11];
  const float* gamma = (const float*)d_in[12];
  const float* beta  = (const float*)d_in[13];

  const int    Btok = NB * LSEQ;                 // 4096 token rows
  const size_t actE = (size_t)Btok * DM;         // activation elements
  const size_t wE   = (size_t)DM * DM;           // weight elements

  // fold 1/sqrt(D_HEAD) and log2(e) into the Q projection (base-2 softmax domain)
  const float SCALE_Q = 0.125f * 1.44269504088896340736f;

  char* ws = (char*)d_ws;
  size_t off = 0;
  auto alloc = [&](size_t bytes) -> char* {
    char* p = ws + off;
    off += (bytes + 255) & ~(size_t)255;
    return p;
  };
  unsigned short* Xq  = (unsigned short*)alloc(actE * 2);
  unsigned short* Xk  = (unsigned short*)alloc(actE * 2);
  unsigned short* Xv  = (unsigned short*)alloc(actE * 2);
  unsigned short* Wqt = (unsigned short*)alloc(wE * 2);
  unsigned short* Wkt = (unsigned short*)alloc(wE * 2);
  unsigned short* Wvt = (unsigned short*)alloc(wE * 2);
  unsigned short* Wot = (unsigned short*)alloc(wE * 2);
  unsigned short* Qb  = (unsigned short*)alloc(actE * 2);
  unsigned short* Kb  = (unsigned short*)alloc(actE * 2);
  unsigned short* Vb  = (unsigned short*)alloc(actE * 2);
  unsigned short* Vtb = (unsigned short*)alloc(actE * 2);
  unsigned short* Ob  = (unsigned short*)alloc(actE * 2);
  float*          Yf  = (float*)alloc(actE * 4);

  const dim3 blk(256);
  const int cvtGrid = (int)(actE / 4 / 256);
  cvt_k<<<cvtGrid, blk, 0, stream>>>(x_q, Xq, (int)actE);
  cvt_k<<<cvtGrid, blk, 0, stream>>>(x_k, Xk, (int)actE);
  cvt_k<<<cvtGrid, blk, 0, stream>>>(x_v, Xv, (int)actE);

  const int wtGrid = (int)((wE + 255) / 256);
  wtcvt_k<<<wtGrid, blk, 0, stream>>>(Wq, Wqt, DM, DM, SCALE_Q);
  wtcvt_k<<<wtGrid, blk, 0, stream>>>(Wk, Wkt, DM, DM, 1.0f);
  wtcvt_k<<<wtGrid, blk, 0, stream>>>(Wv, Wvt, DM, DM, 1.0f);
  wtcvt_k<<<wtGrid, blk, 0, stream>>>(Wo, Wot, DM, DM, 1.0f);

  const dim3 gGrid(DM / 128, Btok / 128);        // (8, 32), 128x128 tiles
  gemm_bf16_k<0><<<gGrid, blk, 0, stream>>>(Xq, Wqt, bq, SCALE_Q, nullptr, Qb, nullptr, Btok, DM, DM);
  gemm_bf16_k<0><<<gGrid, blk, 0, stream>>>(Xk, Wkt, bk, 1.0f,    nullptr, Kb, nullptr, Btok, DM, DM);
  gemm_bf16_k<0><<<gGrid, blk, 0, stream>>>(Xv, Wvt, bv, 1.0f,    nullptr, Vb, nullptr, Btok, DM, DM);

  vtrans_k<<<(int)(actE / 256), blk, 0, stream>>>(Vb, Vtb);

  const dim3 aGrid(LSEQ / 128, NHEADS, NB);      // (16, 16, 2)
  flash_attn_k<<<aGrid, blk, 0, stream>>>(Qb, Kb, Vtb, Ob);

  gemm_bf16_k<1><<<gGrid, blk, 0, stream>>>(Ob, Wot, bo, 1.0f, x_q, nullptr, Yf, Btok, DM, DM);

  ln_k<<<Btok, blk, 0, stream>>>(Yf, gamma, beta, (float*)d_out);
}
